// MOE_Transformer_18872086298684
// MI455X (gfx1250) — compile-verified, matrix operand order checked by reading, not run on hardware
//
#include <hip/hip_runtime.h>
#include <hip/hip_bf16.h>

#define DMODEL 1024
#define SEQ    512
#define NHEAD  16
#define HSZ    64
#define NLAYER 8
#define VOCAB  32000
#define BATCH  4
#define ROWS   (BATCH*SEQ)   /* 2048 */
#define DFF    4096

typedef __bf16 bf16;
typedef __attribute__((ext_vector_type(16))) __bf16 v16bf;
typedef __attribute__((ext_vector_type(8)))  float  v8f;
typedef __attribute__((ext_vector_type(4)))  unsigned int u32x4;

union ABFrag { u32x4 u[2]; v16bf v; };

static __device__ __forceinline__ v8f wmma_bf16(v16bf a, v16bf b, v8f c) {
  // D(f32, 16x16) = A(bf16, 16x32) * B(bf16, 32x16) + C
  return __builtin_amdgcn_wmma_f32_16x16x32_bf16(false, a, false, b, (short)0, c,
                                                 false, false);
}

// ---------------------------------------------------------------------------
// h[row, :] = char_emb[x[row]] + pos_emb[row % T]
__global__ void embed_kernel(const int* __restrict__ x, const float* __restrict__ ce,
                             const float* __restrict__ pe, float* __restrict__ h) {
  int row = blockIdx.x, tid = threadIdx.x;
  int tok = x[row];
  int t = row & (SEQ - 1);
#pragma unroll
  for (int i = 0; i < 4; i++) {
    int c = tid + i * 256;
    h[(size_t)row * DMODEL + c] =
        ce[(size_t)tok * DMODEL + c] + pe[(size_t)t * DMODEL + c];
  }
}

// ---------------------------------------------------------------------------
// LayerNorm over last dim (1024), fp32 in -> bf16 out
__global__ void ln_kernel(const float* __restrict__ x, const float* __restrict__ s,
                          const float* __restrict__ bpar, bf16* __restrict__ y) {
  int row = blockIdx.x, tid = threadIdx.x;
  const float* xr = x + (size_t)row * DMODEL;
  float v0[4]; float sum = 0.f, sq = 0.f;
#pragma unroll
  for (int i = 0; i < 4; i++) {
    float t = xr[tid + i * 256];
    v0[i] = t; sum += t; sq += t * t;
  }
  __shared__ float rs[256], rq[256];
  rs[tid] = sum; rq[tid] = sq; __syncthreads();
  for (int off = 128; off > 0; off >>= 1) {
    if (tid < off) { rs[tid] += rs[tid + off]; rq[tid] += rq[tid + off]; }
    __syncthreads();
  }
  float mean = rs[0] * (1.f / DMODEL);
  float var  = rq[0] * (1.f / DMODEL) - mean * mean;
  float rstd = rsqrtf(var + 1e-5f);
  bf16* yr = y + (size_t)row * DMODEL;
#pragma unroll
  for (int i = 0; i < 4; i++) {
    int c = tid + i * 256;
    yr[c] = (bf16)((v0[i] - mean) * rstd * s[c] + bpar[c]);
  }
}

// ---------------------------------------------------------------------------
// Wt[n, k] = (bf16) W[k, n]   (tiled transpose+convert; K, N multiples of 32)
__global__ void wtrans_kernel(const float* __restrict__ W, bf16* __restrict__ Wt,
                              int K, int N) {
  __shared__ float tile[32][33];
  int n0 = blockIdx.x * 32, k0 = blockIdx.y * 32;
  int tx = threadIdx.x, ty = threadIdx.y;
#pragma unroll
  for (int i = 0; i < 32; i += 8)
    tile[ty + i][tx] = W[(size_t)(k0 + ty + i) * N + n0 + tx];
  __syncthreads();
#pragma unroll
  for (int i = 0; i < 32; i += 8)
    Wt[(size_t)(n0 + ty + i) * K + k0 + tx] = (bf16)tile[tx][ty + i];
}

// ---------------------------------------------------------------------------
// vT[b, h, hs, t] = v[b, t, h*HSZ + hs]
__global__ void vt_kernel(const bf16* __restrict__ v, bf16* __restrict__ vT) {
  int idx = blockIdx.x * 256 + threadIdx.x;
  int t = idx & (SEQ - 1);
  int rest = idx / SEQ;
  int hs = rest & (HSZ - 1);
  int bh = rest / HSZ;
  int b = bh / NHEAD, hh = bh % NHEAD;
  vT[idx] = v[((size_t)(b * SEQ + t)) * DMODEL + hh * HSZ + hs];
}

// ---------------------------------------------------------------------------
// Generic bf16 WMMA GEMM: out = act(A[MxK] * Bt[NxK]^T + bias [+ resid])
// Block = 256 threads = 8 waves (2 row-waves x 4 col-waves), tile 64x128.
// Each wave owns a 32x32 output patch = 2x2 WMMA tiles: 4 WMMAs per 8 b128
// loads per K-step (16 FLOP/B from L2, 1.5x the density of a 2x1 wave tile).
__global__ __launch_bounds__(256) void gemm_bf16_kernel(
    const bf16* __restrict__ A, const bf16* __restrict__ Bt,
    const float* __restrict__ bias, const float* __restrict__ resid,
    float* __restrict__ outF, bf16* __restrict__ outB,
    int M, int N, int K, int relu) {
  int lane = threadIdx.x & 31;
  int w    = threadIdx.x >> 5;
  int hi   = lane >> 4;        // half-wave (0/1)
  int l16  = lane & 15;
  int wr = w >> 2, wc = w & 3;
  int row0 = blockIdx.y * 64  + wr * 32;   // row tiles: row0, row0+16
  int col0 = blockIdx.x * 128 + wc * 32;   // col tiles: col0, col0+16
  const bf16* Ap0 = A  + (size_t)(row0 + l16) * K + hi * 8;
  const bf16* Ap1 = Ap0 + (size_t)16 * K;
  const bf16* Bp0 = Bt + (size_t)(col0 + l16) * K + hi * 8;
  const bf16* Bp1 = Bp0 + (size_t)16 * K;
  v8f acc00 = {}, acc01 = {}, acc10 = {}, acc11 = {};
  for (int k0 = 0; k0 < K; k0 += 32) {
    __builtin_prefetch(Bp0 + k0 + 256, 0, 0);  // global_prefetch_b8
    ABFrag a0, a1, b0, b1;
    a0.u[0] = *(const u32x4*)(Ap0 + k0);
    a0.u[1] = *(const u32x4*)(Ap0 + k0 + 16);
    a1.u[0] = *(const u32x4*)(Ap1 + k0);
    a1.u[1] = *(const u32x4*)(Ap1 + k0 + 16);
    b0.u[0] = *(const u32x4*)(Bp0 + k0);
    b0.u[1] = *(const u32x4*)(Bp0 + k0 + 16);
    b1.u[0] = *(const u32x4*)(Bp1 + k0);
    b1.u[1] = *(const u32x4*)(Bp1 + k0 + 16);
    acc00 = wmma_bf16(a0.v, b0.v, acc00);
    acc01 = wmma_bf16(a0.v, b1.v, acc01);
    acc10 = wmma_bf16(a1.v, b0.v, acc10);
    acc11 = wmma_bf16(a1.v, b1.v, acc11);
  }
  int c0 = col0 + l16, c1 = c0 + 16;
  float bv0 = bias ? bias[c0] : 0.f;
  float bv1 = bias ? bias[c1] : 0.f;
#pragma unroll
  for (int r = 0; r < 8; r++) {
    int ra = row0 + r + 8 * hi;   // C layout: lane holds col l16, rows r + 8*half
    int rb = ra + 16;
    float e00 = acc00[r] + bv0, e01 = acc01[r] + bv1;
    float e10 = acc10[r] + bv0, e11 = acc11[r] + bv1;
    if (resid) {
      e00 += resid[(size_t)ra * N + c0]; e01 += resid[(size_t)ra * N + c1];
      e10 += resid[(size_t)rb * N + c0]; e11 += resid[(size_t)rb * N + c1];
    }
    if (relu) {
      e00 = fmaxf(e00, 0.f); e01 = fmaxf(e01, 0.f);
      e10 = fmaxf(e10, 0.f); e11 = fmaxf(e11, 0.f);
    }
    if (outF) {
      outF[(size_t)ra * N + c0] = e00; outF[(size_t)ra * N + c1] = e01;
      outF[(size_t)rb * N + c0] = e10; outF[(size_t)rb * N + c1] = e11;
    }
    if (outB) {
      outB[(size_t)ra * N + c0] = (bf16)e00; outB[(size_t)ra * N + c1] = (bf16)e01;
      outB[(size_t)rb * N + c0] = (bf16)e10; outB[(size_t)rb * N + c1] = (bf16)e11;
    }
  }
}

// ---------------------------------------------------------------------------
// Flash attention, one block per (b, h). Each wave owns one 16-row query tile,
// online softmax in fp32, scores & P*V via bf16 WMMA. vT is [B,H,HS,T].
__global__ __launch_bounds__(256) void attn_kernel(
    const bf16* __restrict__ q, const bf16* __restrict__ k,
    const bf16* __restrict__ vT, bf16* __restrict__ o) {
  __shared__ __align__(16) bf16 Pl[8][16][40];   // per-wave P staging (padded)
  int bh = blockIdx.x;
  int b = bh / NHEAD, h = bh % NHEAD;
  int lane = threadIdx.x & 31;
  int wid  = threadIdx.x >> 5;
  int hi   = lane >> 4;
  int l16  = lane & 15;

  for (int qt = wid; qt < SEQ / 16; qt += 8) {
    const bf16* qp = q + ((size_t)(b * SEQ + qt * 16 + l16)) * DMODEL + h * HSZ + hi * 8;
    ABFrag qa0, qa1;
    qa0.u[0] = *(const u32x4*)(qp);
    qa0.u[1] = *(const u32x4*)(qp + 16);
    qa1.u[0] = *(const u32x4*)(qp + 32);
    qa1.u[1] = *(const u32x4*)(qp + 48);

    float m_i[8], l_i[8];
    v8f acc[4] = {};
#pragma unroll
    for (int r = 0; r < 8; r++) { m_i[r] = -1e30f; l_i[r] = 0.f; }

    int nch = (16 * qt + 47) >> 5;     // 32-key chunks covering keys <= qt*16+15
    for (int c = 0; c < nch; c++) {
      int kc = c * 32;
      v8f S[2];
#pragma unroll
      for (int st = 0; st < 2; st++) {
        const bf16* kp = k + ((size_t)(b * SEQ + kc + st * 16 + l16)) * DMODEL + h * HSZ + hi * 8;
        ABFrag kb0, kb1;
        kb0.u[0] = *(const u32x4*)(kp);
        kb0.u[1] = *(const u32x4*)(kp + 16);
        kb1.u[0] = *(const u32x4*)(kp + 32);
        kb1.u[1] = *(const u32x4*)(kp + 48);
        v8f sc = {};
        sc = wmma_bf16(qa0.v, kb0.v, sc);
        sc = wmma_bf16(qa1.v, kb1.v, sc);
        S[st] = sc;
      }
      // scale + causal mask + row max
      float rm[8];
#pragma unroll
      for (int r = 0; r < 8; r++) rm[r] = -1e30f;
#pragma unroll
      for (int st = 0; st < 2; st++) {
        int key = kc + st * 16 + l16;
#pragma unroll
        for (int r = 0; r < 8; r++) {
          float sv = S[st][r] * 0.125f;                 // HS^-0.5
          if (key > qt * 16 + r + 8 * hi) sv = -1e30f;  // causal
          S[st][r] = sv;
          rm[r] = fmaxf(rm[r], sv);
        }
      }
#pragma unroll
      for (int r = 0; r < 8; r++) {
#pragma unroll
        for (int off = 1; off < 16; off <<= 1)
          rm[r] = fmaxf(rm[r], __shfl_xor(rm[r], off, 32));
        float mn = fmaxf(m_i[r], rm[r]);
        float alpha = __expf(m_i[r] - mn);
        m_i[r] = mn;
        l_i[r] *= alpha;
#pragma unroll
        for (int nt = 0; nt < 4; nt++) acc[nt][r] *= alpha;
      }
      // P = exp(S - m); stage into LDS in row-major so we can reload as A-frag
      float ps[8];
#pragma unroll
      for (int r = 0; r < 8; r++) ps[r] = 0.f;
#pragma unroll
      for (int st = 0; st < 2; st++)
#pragma unroll
        for (int r = 0; r < 8; r++) {
          float p = __expf(S[st][r] - m_i[r]);
          Pl[wid][r + 8 * hi][st * 16 + l16] = (bf16)p;
          ps[r] += p;
        }
#pragma unroll
      for (int r = 0; r < 8; r++) {
#pragma unroll
        for (int off = 1; off < 16; off <<= 1)
          ps[r] += __shfl_xor(ps[r], off, 32);
        l_i[r] += ps[r];
      }
      // reload P as 16x32 A fragment (wave-internal, no barrier needed)
      ABFrag pa;
      pa.u[0] = *(const u32x4*)&Pl[wid][l16][hi * 8];
      pa.u[1] = *(const u32x4*)&Pl[wid][l16][16 + hi * 8];
#pragma unroll
      for (int nt = 0; nt < 4; nt++) {
        const bf16* vp = vT + ((size_t)bh * HSZ + nt * 16 + l16) * SEQ + kc + hi * 8;
        ABFrag vb;
        vb.u[0] = *(const u32x4*)(vp);
        vb.u[1] = *(const u32x4*)(vp + 16);
        acc[nt] = wmma_bf16(pa.v, vb.v, acc[nt]);
      }
    }
    // normalize and write o[b, t, h*HSZ + hs]
#pragma unroll
    for (int r = 0; r < 8; r++) {
      float inv = 1.f / l_i[r];
      size_t row = (size_t)(b * SEQ + qt * 16 + r + 8 * hi);
#pragma unroll
      for (int nt = 0; nt < 4; nt++)
        o[row * DMODEL + h * HSZ + nt * 16 + l16] = (bf16)(acc[nt][r] * inv);
    }
  }
}

// ---------------------------------------------------------------------------
extern "C" void kernel_launch(void* const* d_in, const int* in_sizes, int n_in,
                              void* d_out, int out_size, void* d_ws, size_t ws_size,
                              hipStream_t stream) {
  (void)in_sizes; (void)n_in; (void)out_size; (void)ws_size;
  const int*   x        = (const int*)d_in[0];
  const float* char_emb = (const float*)d_in[1];
  const float* pos_emb  = (const float*)d_in[2];
  // d_in[3], d_in[4]: router (dead code in reference output) -- skipped
  const float* ln1_s = (const float*)d_in[5];
  const float* ln1_b = (const float*)d_in[6];
  const float* wq = (const float*)d_in[7];
  const float* wk = (const float*)d_in[8];
  const float* wv = (const float*)d_in[9];
  const float* wo = (const float*)d_in[10];
  const float* bo = (const float*)d_in[11];
  const float* ln2_s = (const float*)d_in[12];
  const float* ln2_b = (const float*)d_in[13];
  const float* w1 = (const float*)d_in[14];
  const float* b1 = (const float*)d_in[15];
  const float* w2 = (const float*)d_in[16];
  const float* b2 = (const float*)d_in[17];
  const float* lnf_s = (const float*)d_in[18];
  const float* lnf_b = (const float*)d_in[19];
  const float* wf  = (const float*)d_in[20];
  const float* bfv = (const float*)d_in[21];
  float* out = (float*)d_out;

  // workspace layout (bytes)
  char* ws = (char*)d_ws;
  float* h    = (float*)(ws + 0);          //  8 MB fp32 residual stream
  bf16* y     = (bf16*)(ws + 8388608);     //  4 MB LN output (reused per stage)
  bf16* qb    = (bf16*)(ws + 12582912);    //  4 MB
  bf16* kb    = (bf16*)(ws + 16777216);    //  4 MB
  bf16* vb    = (bf16*)(ws + 20971520);    //  4 MB
  bf16* vT    = (bf16*)(ws + 25165824);    //  4 MB
  bf16* ob    = (bf16*)(ws + 29360128);    //  4 MB
  bf16* ffn1  = (bf16*)(ws + 33554432);    // 16 MB
  bf16* wqT   = (bf16*)(ws + 50331648);    //  2 MB each
  bf16* wkT   = (bf16*)(ws + 52428800);
  bf16* wvT   = (bf16*)(ws + 54525952);
  bf16* woT   = (bf16*)(ws + 56623104);
  bf16* w1T   = (bf16*)(ws + 58720256);    //  8 MB
  bf16* w2T   = (bf16*)(ws + 67108864);    //  8 MB
  bf16* wfT   = (bf16*)(ws + 75497472);    // 65.5 MB  (end ~141 MB)

  dim3 tb(32, 8);

  embed_kernel<<<ROWS, 256, 0, stream>>>(x, char_emb, pos_emb, h);
  wtrans_kernel<<<dim3(VOCAB / 32, DMODEL / 32), tb, 0, stream>>>(wf, wfT, DMODEL, VOCAB);

  for (int l = 0; l < NLAYER; l++) {
    ln_kernel<<<ROWS, 256, 0, stream>>>(h, ln1_s + l * DMODEL, ln1_b + l * DMODEL, y);

    size_t wOff = (size_t)l * DMODEL * DMODEL;
    wtrans_kernel<<<dim3(32, 32), tb, 0, stream>>>(wq + wOff, wqT, DMODEL, DMODEL);
    wtrans_kernel<<<dim3(32, 32), tb, 0, stream>>>(wk + wOff, wkT, DMODEL, DMODEL);
    wtrans_kernel<<<dim3(32, 32), tb, 0, stream>>>(wv + wOff, wvT, DMODEL, DMODEL);
    wtrans_kernel<<<dim3(32, 32), tb, 0, stream>>>(wo + wOff, woT, DMODEL, DMODEL);

    gemm_bf16_kernel<<<dim3(DMODEL / 128, ROWS / 64), 256, 0, stream>>>(
        y, wqT, nullptr, nullptr, nullptr, qb, ROWS, DMODEL, DMODEL, 0);
    gemm_bf16_kernel<<<dim3(DMODEL / 128, ROWS / 64), 256, 0, stream>>>(
        y, wkT, nullptr, nullptr, nullptr, kb, ROWS, DMODEL, DMODEL, 0);
    gemm_bf16_kernel<<<dim3(DMODEL / 128, ROWS / 64), 256, 0, stream>>>(
        y, wvT, nullptr, nullptr, nullptr, vb, ROWS, DMODEL, DMODEL, 0);

    vt_kernel<<<(BATCH * NHEAD * HSZ * SEQ) / 256, 256, 0, stream>>>(vb, vT);
    attn_kernel<<<BATCH * NHEAD, 256, 0, stream>>>(qb, kb, vT, ob);

    gemm_bf16_kernel<<<dim3(DMODEL / 128, ROWS / 64), 256, 0, stream>>>(
        ob, woT, bo + l * DMODEL, h, h, nullptr, ROWS, DMODEL, DMODEL, 0);

    ln_kernel<<<ROWS, 256, 0, stream>>>(h, ln2_s + l * DMODEL, ln2_b + l * DMODEL, y);

    wtrans_kernel<<<dim3(DFF / 32, DMODEL / 32), tb, 0, stream>>>(
        w1 + (size_t)l * DMODEL * DFF, w1T, DMODEL, DFF);
    gemm_bf16_kernel<<<dim3(DFF / 128, ROWS / 64), 256, 0, stream>>>(
        y, w1T, b1 + l * DFF, nullptr, nullptr, ffn1, ROWS, DFF, DMODEL, 1);

    wtrans_kernel<<<dim3(DMODEL / 32, DFF / 32), tb, 0, stream>>>(
        w2 + (size_t)l * DFF * DMODEL, w2T, DFF, DMODEL);
    gemm_bf16_kernel<<<dim3(DMODEL / 128, ROWS / 64), 256, 0, stream>>>(
        ffn1, w2T, b2 + l * DMODEL, h, h, nullptr, ROWS, DMODEL, DFF, 0);
  }

  ln_kernel<<<ROWS, 256, 0, stream>>>(h, lnf_s, lnf_b, y);
  gemm_bf16_kernel<<<dim3(VOCAB / 128, ROWS / 64), 256, 0, stream>>>(
      y, wfT, bfv, nullptr, out, nullptr, ROWS, VOCAB, DMODEL, 0);
}
